// JaxFusedMoE_75600014344554
// MI455X (gfx1250) — matrix operand order repaired and checked by reading, not compiled, not run on hardware
//
#include <hip/hip_runtime.h>
#include <hip/hip_bf16.h>
#include <math.h>

#define USE_ASYNC_LDS 1

typedef __attribute__((ext_vector_type(16))) _Float16 v16h;
typedef __attribute__((ext_vector_type(8)))  _Float16 v8h;
typedef __attribute__((ext_vector_type(4)))  _Float16 v4h;
typedef __attribute__((ext_vector_type(8)))  float    v8f;

namespace {
constexpr int T = 2048;   // tokens
constexpr int H = 1024;   // hidden
constexpr int I = 2048;   // intermediate
constexpr int E = 8;      // experts
constexpr int CAP = 2048; // max pairs per expert
constexpr int MT = 64;    // M tile per block (4 WMMA sub-tiles per wave)
constexpr int APITCH = 40; // LDS A row pitch in halfs (80B, 16B-aligned, bank-spread)
} // namespace

__device__ __forceinline__ v8f wmma16x16x32(v16h a, v16h b, v8f c) {
  return __builtin_amdgcn_wmma_f32_16x16x32_f16(
      false, a, false, b, (short)0, c, false, false);
}

// A operand from an LDS row (current 32-wide K chunk).
__device__ __forceinline__ v16h lda_lds(const _Float16* row, int khalf) {
  v8h lo = *reinterpret_cast<const v8h*>(row + khalf * 8);
  v8h hi = *reinterpret_cast<const v8h*>(row + 16 + khalf * 8);
  v16h r;
#pragma unroll
  for (int i = 0; i < 8; ++i) { r[i] = lo[i]; r[8 + i] = hi[i]; }
  return r;
}

// B operand 32x16 (KxN), weights N-major f16.
__device__ __forceinline__ v16h ldb_f16(const _Float16* colk, int khalf) {
  v8h lo = *reinterpret_cast<const v8h*>(colk + khalf * 8);
  v8h hi = *reinterpret_cast<const v8h*>(colk + 16 + khalf * 8);
  v16h r;
#pragma unroll
  for (int i = 0; i < 8; ++i) { r[i] = lo[i]; r[8 + i] = hi[i]; }
  return r;
}

#if USE_ASYNC_LDS
// Async DMA: 16 bytes global -> LDS, tracked by ASYNCcnt (no VGPR round trip).
__device__ __forceinline__ void async_copy_b128(unsigned lds_addr, const void* gptr) {
  asm volatile("global_load_async_to_lds_b128 %0, %1, off"
               :: "v"(lds_addr), "v"((unsigned long long)(uintptr_t)gptr)
               : "memory");
}
__device__ __forceinline__ void wait_async0() {
  asm volatile("s_wait_asynccnt 0" ::: "memory");
}
#endif

// ---------------------------------------------------------------------------
// 0) f32 -> f16 streaming conversion (weights + hidden), done once.
// ---------------------------------------------------------------------------
__global__ __launch_bounds__(256) void cvt_f32_f16_kernel(
    const float* __restrict__ in, _Float16* __restrict__ out, int n4) {
  int i = blockIdx.x * 256 + threadIdx.x;
  if (i >= n4) return;
  float4 x = reinterpret_cast<const float4*>(in)[i];
  v4h h;
  h[0] = (_Float16)x.x; h[1] = (_Float16)x.y;
  h[2] = (_Float16)x.z; h[3] = (_Float16)x.w;
  reinterpret_cast<v4h*>(out)[i] = h;
}

// ---------------------------------------------------------------------------
// 1) Routing: softmax -> top-2 -> renormalize.
// ---------------------------------------------------------------------------
__global__ __launch_bounds__(256) void routing_kernel(
    const float* __restrict__ gating, int* __restrict__ topk_idx,
    float* __restrict__ topk_w) {
  int t = blockIdx.x * blockDim.x + threadIdx.x;
  if (t >= T) return;
  float g[E];
#pragma unroll
  for (int e = 0; e < E; ++e) g[e] = gating[t * E + e];
  int a = 0; float best = g[0];
#pragma unroll
  for (int e = 1; e < E; ++e) if (g[e] > best) { best = g[e]; a = e; }
  int b = -1; float second = -INFINITY;
#pragma unroll
  for (int e = 0; e < E; ++e)
    if (e != a && g[e] > second) { second = g[e]; b = e; }
  float eb = expf(second - best);
  float inv = 1.0f / (1.0f + eb);
  topk_idx[2 * t + 0] = a;  topk_w[2 * t + 0] = inv;
  topk_idx[2 * t + 1] = b;  topk_w[2 * t + 1] = eb * inv;
}

__global__ void init_counts_kernel(int* __restrict__ counts) {
  if (threadIdx.x < E) counts[threadIdx.x] = 0;
}

__global__ __launch_bounds__(256) void scatter_kernel(
    const int* __restrict__ topk_idx, int* __restrict__ counts,
    int* __restrict__ lists) {
  int t = blockIdx.x * blockDim.x + threadIdx.x;
  if (t >= T) return;
#pragma unroll
  for (int k = 0; k < 2; ++k) {
    int p = 2 * t + k;
    int e = topk_idx[p];
    int pos = atomicAdd(&counts[e], 1);
    lists[e * CAP + pos] = p;
  }
}

// ---------------------------------------------------------------------------
// 3) GEMM1 + SwiGLU.  Block = 128 threads (4 waves).
//    Block tile: M=64 pairs x (64 gate + matching 64 up cols).
//    Wave: 4 M-subtiles x (1 gate + 1 up) = 8 accumulators; each B tile feeds
//    4 WMMAs.  A tile async-DMA'd into LDS (double-buffered); B tiles
//    software-pipelined one k-step ahead in registers.
// ---------------------------------------------------------------------------
__global__ __launch_bounds__(128) void gemm1_swiglu_kernel(
    const _Float16* __restrict__ hh, const _Float16* __restrict__ w1h,
    const int* __restrict__ counts, const int* __restrict__ lists,
    _Float16* __restrict__ act) {
  const int e = blockIdx.z;
  const int cnt = counts[e];
  const int mbase = blockIdx.y * MT;
  if (mbase >= cnt) return;                     // block-uniform

  __shared__ __align__(16) _Float16 la[2 * MT * APITCH];

  const int tid   = threadIdx.x;
  const int wave  = tid >> 5;
  const int lane  = tid & 31;
  const int khalf = lane >> 4;
  const int nlane = lane & 15;
  const int gcol  = blockIdx.x * 64 + wave * 16 + nlane;   // gate column

  // Stager: thread copies halfs [spair, spair+16) of row srow (2 x b128).
  const int srow  = tid >> 1;
  const int spair = (tid & 1) * 16;
  int sgm = mbase + srow;
  int sp  = lists[e * CAP + (sgm < cnt ? sgm : cnt - 1)];
  const _Float16* ssrc = hh + (size_t)(sp >> 1) * H + spair;
#if USE_ASYNC_LDS
  const unsigned lbase =
      (unsigned)(uintptr_t)&la[0] + (unsigned)(srow * APITCH + spair) * 2u;
#else
  _Float16* sdst0 = &la[srow * APITCH + spair];
#endif

  const _Float16* gptr = w1h + ((size_t)e * (2 * I) + gcol) * H;
  const _Float16* uptr = gptr + (size_t)I * H;

  v8f accg[4] = {}, accu[4] = {};

  // prologue: stage A chunk 0, and pre-load B for k=0
#if USE_ASYNC_LDS
  async_copy_b128(lbase, ssrc);
  async_copy_b128(lbase + 16, ssrc + 8);
#else
  *reinterpret_cast<v8h*>(sdst0) = *reinterpret_cast<const v8h*>(ssrc);
  *reinterpret_cast<v8h*>(sdst0 + 8) = *reinterpret_cast<const v8h*>(ssrc + 8);
#endif
  v16h Bg = ldb_f16(gptr, khalf);
  v16h Bu = ldb_f16(uptr, khalf);

  int cur = 0;
  for (int k = 0; k < H; k += 32) {
#if USE_ASYNC_LDS
    wait_async0();
#endif
    __syncthreads();
    if (k + 32 < H) {
      const _Float16* s = ssrc + k + 32;
#if USE_ASYNC_LDS
      unsigned d = lbase + (unsigned)((cur ^ 1) * MT * APITCH) * 2u;
      async_copy_b128(d, s);
      async_copy_b128(d + 16, s + 8);
#else
      _Float16* d = sdst0 + (cur ^ 1) * MT * APITCH;
      *reinterpret_cast<v8h*>(d) = *reinterpret_cast<const v8h*>(s);
      *reinterpret_cast<v8h*>(d + 8) = *reinterpret_cast<const v8h*>(s + 8);
#endif
    }
    const _Float16* abase = &la[cur * MT * APITCH];
    v16h A0 = lda_lds(abase + ((lane & 15) + 0)  * APITCH, khalf);
    v16h A1 = lda_lds(abase + ((lane & 15) + 16) * APITCH, khalf);
    v16h A2 = lda_lds(abase + ((lane & 15) + 32) * APITCH, khalf);
    v16h A3 = lda_lds(abase + ((lane & 15) + 48) * APITCH, khalf);
    // next-iteration B (branchless clamp keeps last iteration in bounds)
    int kn = (k + 32 < H) ? (k + 32) : k;
    __builtin_prefetch(gptr + k + 128, 0, 3);
    v16h BgN = ldb_f16(gptr + kn, khalf);
    v16h BuN = ldb_f16(uptr + kn, khalf);
    accg[0] = wmma16x16x32(A0, Bg, accg[0]);
    accg[1] = wmma16x16x32(A1, Bg, accg[1]);
    accg[2] = wmma16x16x32(A2, Bg, accg[2]);
    accg[3] = wmma16x16x32(A3, Bg, accg[3]);
    accu[0] = wmma16x16x32(A0, Bu, accu[0]);
    accu[1] = wmma16x16x32(A1, Bu, accu[1]);
    accu[2] = wmma16x16x32(A2, Bu, accu[2]);
    accu[3] = wmma16x16x32(A3, Bu, accu[3]);
    Bg = BgN;
    Bu = BuN;
    cur ^= 1;
  }

  // D layout: vgpr v at lane L -> M = v + 8*(L>>4), N = L&15.
#pragma unroll
  for (int mt = 0; mt < 4; ++mt) {
#pragma unroll
    for (int v = 0; v < 8; ++v) {
      int gm = mbase + mt * 16 + v + khalf * 8;
      if (gm < cnt) {
        int p = lists[e * CAP + gm];
        float g = accg[mt][v];
        float u = accu[mt][v];
        float s = g / (1.0f + expf(-g));       // silu
        act[(size_t)p * I + gcol] = (_Float16)(s * u);
      }
    }
  }
}

// ---------------------------------------------------------------------------
// 4) GEMM2: out2[p, :H] = act[p, :I] . w2[e]^T.  M=64 x N=128 per block;
//    wave = 4 M-subtiles x 2 N-tiles = 8 accumulators; B pipelined.
// ---------------------------------------------------------------------------
__global__ __launch_bounds__(128) void gemm2_kernel(
    const _Float16* __restrict__ act, const _Float16* __restrict__ w2h,
    const int* __restrict__ counts, const int* __restrict__ lists,
    float* __restrict__ out2) {
  const int e = blockIdx.z;
  const int cnt = counts[e];
  const int mbase = blockIdx.y * MT;
  if (mbase >= cnt) return;

  __shared__ __align__(16) _Float16 la[2 * MT * APITCH];

  const int tid   = threadIdx.x;
  const int wave  = tid >> 5;
  const int lane  = tid & 31;
  const int khalf = lane >> 4;
  const int nlane = lane & 15;
  const int nbase = blockIdx.x * 128 + wave * 32;

  const int srow  = tid >> 1;
  const int spair = (tid & 1) * 16;
  int sgm = mbase + srow;
  int sp  = lists[e * CAP + (sgm < cnt ? sgm : cnt - 1)];
  const _Float16* ssrc = act + (size_t)sp * I + spair;
#if USE_ASYNC_LDS
  const unsigned lbase =
      (unsigned)(uintptr_t)&la[0] + (unsigned)(srow * APITCH + spair) * 2u;
#else
  _Float16* sdst0 = &la[srow * APITCH + spair];
#endif

  const _Float16* bptr[2];
#pragma unroll
  for (int nt = 0; nt < 2; ++nt)
    bptr[nt] = w2h + ((size_t)e * H + (nbase + nt * 16 + nlane)) * I;

  v8f acc[4][2] = {};

#if USE_ASYNC_LDS
  async_copy_b128(lbase, ssrc);
  async_copy_b128(lbase + 16, ssrc + 8);
#else
  *reinterpret_cast<v8h*>(sdst0) = *reinterpret_cast<const v8h*>(ssrc);
  *reinterpret_cast<v8h*>(sdst0 + 8) = *reinterpret_cast<const v8h*>(ssrc + 8);
#endif
  v16h B0 = ldb_f16(bptr[0], khalf);
  v16h B1 = ldb_f16(bptr[1], khalf);

  int cur = 0;
  for (int k = 0; k < I; k += 32) {
#if USE_ASYNC_LDS
    wait_async0();
#endif
    __syncthreads();
    if (k + 32 < I) {
      const _Float16* s = ssrc + k + 32;
#if USE_ASYNC_LDS
      unsigned d = lbase + (unsigned)((cur ^ 1) * MT * APITCH) * 2u;
      async_copy_b128(d, s);
      async_copy_b128(d + 16, s + 8);
#else
      _Float16* d = sdst0 + (cur ^ 1) * MT * APITCH;
      *reinterpret_cast<v8h*>(d) = *reinterpret_cast<const v8h*>(s);
      *reinterpret_cast<v8h*>(d + 8) = *reinterpret_cast<const v8h*>(s + 8);
#endif
    }
    const _Float16* abase = &la[cur * MT * APITCH];
    v16h A0 = lda_lds(abase + ((lane & 15) + 0)  * APITCH, khalf);
    v16h A1 = lda_lds(abase + ((lane & 15) + 16) * APITCH, khalf);
    v16h A2 = lda_lds(abase + ((lane & 15) + 32) * APITCH, khalf);
    v16h A3 = lda_lds(abase + ((lane & 15) + 48) * APITCH, khalf);
    int kn = (k + 32 < I) ? (k + 32) : k;
    __builtin_prefetch(bptr[0] + k + 128, 0, 3);
    v16h B0N = ldb_f16(bptr[0] + kn, khalf);
    v16h B1N = ldb_f16(bptr[1] + kn, khalf);
    acc[0][0] = wmma16x16x32(A0, B0, acc[0][0]);
    acc[1][0] = wmma16x16x32(A1, B0, acc[1][0]);
    acc[2][0] = wmma16x16x32(A2, B0, acc[2][0]);
    acc[3][0] = wmma16x16x32(A3, B0, acc[3][0]);
    acc[0][1] = wmma16x16x32(A0, B1, acc[0][1]);
    acc[1][1] = wmma16x16x32(A1, B1, acc[1][1]);
    acc[2][1] = wmma16x16x32(A2, B1, acc[2][1]);
    acc[3][1] = wmma16x16x32(A3, B1, acc[3][1]);
    B0 = B0N;
    B1 = B1N;
    cur ^= 1;
  }

#pragma unroll
  for (int mt = 0; mt < 4; ++mt) {
#pragma unroll
    for (int v = 0; v < 8; ++v) {
      int gm = mbase + mt * 16 + v + khalf * 8;
      if (gm < cnt) {
        int p = lists[e * CAP + gm];
#pragma unroll
        for (int nt = 0; nt < 2; ++nt)
          out2[(size_t)p * H + (nbase + nt * 16 + nlane)] = acc[mt][nt][v];
      }
    }
  }
}

// ---------------------------------------------------------------------------
// 5) Weighted combine: out[t] = w0 * out2[2t] + w1 * out2[2t+1]
// ---------------------------------------------------------------------------
__global__ __launch_bounds__(256) void combine_kernel(
    const float* __restrict__ out2, const float* __restrict__ topk_w,
    float* __restrict__ out) {
  int i = blockIdx.x * blockDim.x + threadIdx.x;
  if (i >= T * H / 4) return;
  int t = i / (H / 4);
  int c = i % (H / 4);
  const float4* r0 = reinterpret_cast<const float4*>(out2) + (size_t)(2 * t) * (H / 4) + c;
  const float4* r1 = reinterpret_cast<const float4*>(out2) + (size_t)(2 * t + 1) * (H / 4) + c;
  float4 a = *r0, b = *r1;
  float w0 = topk_w[2 * t], w1 = topk_w[2 * t + 1];
  float4 r;
  r.x = w0 * a.x + w1 * b.x;
  r.y = w0 * a.y + w1 * b.y;
  r.z = w0 * a.z + w1 * b.z;
  r.w = w0 * a.w + w1 * b.w;
  reinterpret_cast<float4*>(out)[i] = r;
}

// ---------------------------------------------------------------------------
extern "C" void kernel_launch(void* const* d_in, const int* in_sizes, int n_in,
                              void* d_out, int out_size, void* d_ws, size_t ws_size,
                              hipStream_t stream) {
  const float* hidden = (const float*)d_in[0];   // [T, H]
  const float* w1     = (const float*)d_in[1];   // [E, 2I, H]
  const float* w2     = (const float*)d_in[2];   // [E, H, I]
  const float* gating = (const float*)d_in[3];   // [T, E]
  float* out = (float*)d_out;                    // [T, H]

  char* ws = (char*)d_ws;
  size_t off = 0;
  int*      topk_idx = (int*)(ws + off);       off += 16384;
  float*    topk_w   = (float*)(ws + off);     off += 16384;
  int*      counts   = (int*)(ws + off);       off += 256;
  int*      lists    = (int*)(ws + off);       off += (size_t)E * CAP * 4;        // 64 KiB
  _Float16* act      = (_Float16*)(ws + off);  off += (size_t)2 * T * I * 2;     // 16 MiB
  float*    out2     = (float*)(ws + off);     off += (size_t)2 * T * H * 4;     // 16 MiB
  _Float16* hh       = (_Float16*)(ws + off);  off += (size_t)T * H * 2;         //  4 MiB
  _Float16* w1h      = (_Float16*)(ws + off);  off += (size_t)E * 2 * I * H * 2; // 64 MiB
  _Float16* w2h      = (_Float16*)(ws + off);  off += (size_t)E * H * I * 2;     // 32 MiB
  (void)ws_size; (void)in_sizes; (void)n_in; (void)out_size; (void)off;

  {
    int n4 = T * H / 4;
    cvt_f32_f16_kernel<<<(n4 + 255) / 256, 256, 0, stream>>>(hidden, hh, n4);
  }
  {
    int n4 = E * 2 * I * H / 4;
    cvt_f32_f16_kernel<<<(n4 + 255) / 256, 256, 0, stream>>>(w1, w1h, n4);
  }
  {
    int n4 = E * H * I / 4;
    cvt_f32_f16_kernel<<<(n4 + 255) / 256, 256, 0, stream>>>(w2, w2h, n4);
  }

  routing_kernel<<<T / 256, 256, 0, stream>>>(gating, topk_idx, topk_w);
  init_counts_kernel<<<1, 32, 0, stream>>>(counts);
  scatter_kernel<<<T / 256, 256, 0, stream>>>(topk_idx, counts, lists);

  {
    dim3 grid(I / 64, CAP / MT, E);    // 32 x 32 x 8, dead tiles exit early
    gemm1_swiglu_kernel<<<grid, 128, 0, stream>>>(hh, w1h, counts, lists, act);
  }
  {
    dim3 grid(H / 128, CAP / MT, E);   // 8 x 32 x 8
    gemm2_kernel<<<grid, 128, 0, stream>>>(act, w2h, counts, lists, out2);
  }
  combine_kernel<<<(T * H / 4 + 255) / 256, 256, 0, stream>>>(out2, topk_w, out);
}
